// PostProcessHOI_31842887533360
// MI455X (gfx1250) — matrix operand order repaired and checked by reading, not compiled, not run on hardware
//
#include <hip/hip_runtime.h>
#include <hip/hip_bf16.h>
#include <math.h>

// ---------------- problem constants (from reference) ----------------
#define B_N 256
#define Q_N 300
#define C_N 81
#define V_N 117
#define K_N 100
#define QC_N (Q_N * C_N)        // 24300
#define NMS_TH 0.5f
#define CPAD 96                 // C padded to 3 x 32 (WMMA K dim)
#define VPAD 128                // V padded to 8 x 16

// d_out layout (floats), concatenated in reference return order
#define OFF_FS 0ull                                   // final_scores B*K*V
#define OFF_TV (OFF_FS + (size_t)B_N * K_N * V_N)     // topk_values  B*K
#define OFF_LB (OFF_TV + (size_t)B_N * K_N)           // obj_labels   B*K
#define OFF_SB (OFF_LB + (size_t)B_N * K_N)           // sub_boxes    B*K*4
#define OFF_OB (OFF_SB + (size_t)B_N * K_N * 4)       // obj_boxes    B*K*4
#define OFF_KP (OFF_OB + (size_t)B_N * K_N * 4)       // keep         B*K

typedef __attribute__((ext_vector_type(16))) _Float16 v16h;
typedef __attribute__((ext_vector_type(8)))  float    v8f;

// gfx1250 async global->LDS copy path (ASYNCcnt). Probe result: builtins take
// (payload-typed global ptr, payload-typed LDS ptr, imm offset, imm cpol):
//   b128 -> int4* params, b32 -> int* params (device/LDS address spaces).
#if defined(__AMDGCN__) && \
    __has_builtin(__builtin_amdgcn_global_load_async_to_lds_b128) && \
    __has_builtin(__builtin_amdgcn_global_load_async_to_lds_b32) && \
    __has_builtin(__builtin_amdgcn_s_wait_asynccnt)
#define USE_ASYNC 1
typedef int v4i_t __attribute__((ext_vector_type(4)));
typedef __attribute__((address_space(1))) v4i_t* gptr_b128;
typedef __attribute__((address_space(3))) v4i_t* lptr_b128;
typedef __attribute__((address_space(1))) int*   gptr_b32;
typedef __attribute__((address_space(3))) int*   lptr_b32;
#else
#define USE_ASYNC 0
#endif

// =====================================================================
// Kernel 1: per-batch softmax (LDS-resident) + iterative top-100 + boxes
// One workgroup (1024 threads = 32 waves) per batch image.
// Dynamic LDS: 24300 floats of probs + 1024 vals + 1024 idx (~105 KB / 320 KB)
// =====================================================================
__global__ __launch_bounds__(1024)
void hoi_softmax_topk_kernel(const float* __restrict__ obj_logits,
                             const float* __restrict__ sub_boxes,
                             const float* __restrict__ obj_boxes,
                             const int*   __restrict__ tsizes,
                             float* __restrict__ out,
                             int* __restrict__ ws_q,
                             int* __restrict__ ws_lab)
{
    extern __shared__ float smem[];
    float* prob = smem;                 // QC_N floats
    float* rv   = prob + QC_N;          // 1024 floats
    int*   ri   = (int*)(rv + 1024);    // 1024 ints

    const int b   = blockIdx.x;
    const int tid = threadIdx.x;
    const float* src = obj_logits + (size_t)b * QC_N;

    // Stage the whole batch row of logits into LDS: one HBM pass.
#if USE_ASYNC
    // 16B per lane, async DMA into LDS (QC_N is a multiple of 4 floats).
    for (int i = tid * 4; i < QC_N; i += 1024 * 4) {
        __builtin_amdgcn_global_load_async_to_lds_b128(
            (gptr_b128)(src + i), (lptr_b128)(prob + i), 0, 0);
    }
    __builtin_amdgcn_s_wait_asynccnt(0);
#else
    for (int i = tid; i < QC_N; i += 1024) {
        __builtin_prefetch(src + i + 4096, 0, 0);   // global_prefetch_b8
        prob[i] = src[i];
    }
#endif
    __syncthreads();

    // Per-row softmax; each wave32 owns rows q = wave, wave+32, ...
    const int wave = tid >> 5;
    const int lane = tid & 31;
    for (int q = wave; q < Q_N; q += 32) {
        float* row = prob + q * C_N;
        float mx = -3.4e38f;
        for (int e = lane; e < C_N; e += 32) mx = fmaxf(mx, row[e]);
        #pragma unroll
        for (int o = 16; o > 0; o >>= 1) mx = fmaxf(mx, __shfl_xor(mx, o, 32));
        float s = 0.f;
        for (int e = lane; e < C_N; e += 32) {
            float t = expf(row[e] - mx);
            row[e] = t;
            s += t;
        }
        #pragma unroll
        for (int o = 16; o > 0; o >>= 1) s += __shfl_xor(s, o, 32);
        float inv = 1.f / s;
        for (int e = lane; e < C_N; e += 32) row[e] *= inv;
    }
    __syncthreads();

    float* out_tv = out + OFF_TV;
    float* out_lb = out + OFF_LB;

    // 100 rounds of argmax extraction, all in LDS (probs >= 0, -1 = removed).
    // Tie-break: smaller flat index first (matches lax.top_k).
    for (int t = 0; t < K_N; ++t) {
        float bv = -1.f; int bi = QC_N;
        for (int i = tid; i < QC_N; i += 1024) {
            float v = prob[i];
            if (v > bv) { bv = v; bi = i; }
        }
        rv[tid] = bv; ri[tid] = bi;
        __syncthreads();
        for (int s2 = 512; s2 > 0; s2 >>= 1) {
            if (tid < s2) {
                float ov = rv[tid + s2]; int oi = ri[tid + s2];
                if (ov > rv[tid] || (ov == rv[tid] && oi < ri[tid])) {
                    rv[tid] = ov; ri[tid] = oi;
                }
            }
            __syncthreads();
        }
        if (tid == 0) {
            int idx = ri[0];
            prob[idx] = -1.f;
            out_tv[b * K_N + t] = rv[0];
            int lab = idx % C_N;
            int qi  = idx / C_N;
            out_lb[b * K_N + t] = (float)lab;
            ws_lab[b * K_N + t] = lab;
            ws_q[b * K_N + t]   = qi;
        }
        __syncthreads();
    }

    // Gather + cxcywh->xyxy + scale boxes for the 100 winners.
    if (tid < K_N) {
        int qi = ws_q[b * K_N + tid];
        float W = (float)tsizes[b * 2 + 1];
        float H = (float)tsizes[b * 2 + 0];
        const float* sb = sub_boxes + ((size_t)b * Q_N + qi) * 4;
        const float* ob = obj_boxes + ((size_t)b * Q_N + qi) * 4;
        float* osb = out + OFF_SB + ((size_t)b * K_N + tid) * 4;
        float* oob = out + OFF_OB + ((size_t)b * K_N + tid) * 4;
        {
            float cx = sb[0], cy = sb[1], w = sb[2], h = sb[3];
            osb[0] = (cx - 0.5f * w) * W;
            osb[1] = (cy - 0.5f * h) * H;
            osb[2] = (cx + 0.5f * w) * W;
            osb[3] = (cy + 0.5f * h) * H;
        }
        {
            float cx = ob[0], cy = ob[1], w = ob[2], h = ob[3];
            oob[0] = (cx - 0.5f * w) * W;
            oob[1] = (cy - 0.5f * h) * H;
            oob[2] = (cx + 0.5f * w) * W;
            oob[3] = (cy + 0.5f * h) * H;
        }
    }
}

// =====================================================================
// Kernel 2: masks via WMMA (gather-as-GEMM, exact with 0/1 operands),
// fused with sigmoid(verb_logits) * topk_value.
//   masks[k,v] = sum_c onehot(label_k)[c] * correct_mat[v,c]
// One block (256 thr = 8 waves) per batch; wave w owns v-tile w and loops
// the 7 k-tiles. correct_mat is staged ONCE into zero-padded f16 LDS so
// B-fragments are two aligned ds_load_b128 reads (no exec-branchy VMEM).
// =====================================================================
__global__ __launch_bounds__(256)
void hoi_mask_score_kernel(const float* __restrict__ verb_logits,
                           const float* __restrict__ correct_mat,
                           const int*   __restrict__ ws_q,
                           const int*   __restrict__ ws_lab,
                           float* __restrict__ out)
{
    __shared__ __align__(32) _Float16 Bt[VPAD * CPAD];   // 24576 B, Bt[v][c]
#if USE_ASYNC
    __shared__ float raw[V_N * C_N];                     // 37908 B
#endif
    const int b   = blockIdx.x;
    const int tid = threadIdx.x;

#if USE_ASYNC
    for (int i = tid; i < V_N * C_N; i += 256) {
        __builtin_amdgcn_global_load_async_to_lds_b32(
            (gptr_b32)(correct_mat + i), (lptr_b32)(raw + i), 0, 0);
    }
    __builtin_amdgcn_s_wait_asynccnt(0);
    __syncthreads();
    for (int idx = tid; idx < VPAD * CPAD; idx += 256) {
        int v = idx / CPAD, c = idx % CPAD;
        float m = (v < V_N && c < C_N) ? raw[v * C_N + c] : 0.f;
        Bt[idx] = (_Float16)m;
    }
#else
    for (int idx = tid; idx < VPAD * CPAD; idx += 256) {
        int v = idx / CPAD, c = idx % CPAD;
        float m = (v < V_N && c < C_N) ? correct_mat[(size_t)v * C_N + c] : 0.f;
        Bt[idx] = (_Float16)m;
    }
#endif
    __syncthreads();

    const int vt   = tid >> 5;        // wave id == v-tile (0..7)
    const int lane = tid & 31;
    const int half = lane >> 4;       // 0: lanes 0-15, 1: lanes 16-31
    const int lm   = lane & 15;
    const int v    = vt * 16 + lm;    // N index (column of D / row of Bt)

    const float* out_tv = out + OFF_TV;
    float* out_fs = out + OFF_FS;

    for (int kt = 0; kt < 7; ++kt) {
        // A-matrix row (M) = lane%16 for both lane halves (ISA 7.12.2).
        const int krow = kt * 16 + lm;
        const int lab  = (krow < K_N) ? ws_lab[b * K_N + krow] : -1;

        union { v16h v; _Float16 h[16]; } A, Bm;
        union { v8f  v; float    f[8];  } acc;
        #pragma unroll
        for (int r = 0; r < 8; ++r) acc.f[r] = 0.f;

        #pragma unroll
        for (int kb = 0; kb < 3; ++kb) {
            // A fragment: contiguous K order (matches B) -> K = e + 16*half
            #pragma unroll
            for (int e = 0; e < 16; ++e) {
                int kk = e + (half << 4);
                int c  = kb * 32 + kk;
                A.h[e] = (_Float16)((c == lab) ? 1.0f : 0.0f);
            }
            // B fragment: lanes 0-15 K=0..15, lanes 16-31 K=16..31, N=lane%16.
            // Contiguous 16 halfs in LDS (32B, 16B-aligned): two ds_load_b128.
            Bm.v = *(const v16h*)&Bt[v * CPAD + kb * 32 + (half << 4)];

            acc.v = __builtin_amdgcn_wmma_f32_16x16x32_f16(
                /*neg_a=*/false, A.v, /*neg_b=*/false, Bm.v,
                /*c_mod=*/(short)0, acc.v, /*reuse_a=*/false, /*reuse_b=*/false);
        }

        // D layout: VGPR r -> M = 8*half + r, N = lane%16.
        #pragma unroll
        for (int r = 0; r < 8; ++r) {
            int k = kt * 16 + (half << 3) + r;
            if (k < K_N && v < V_N) {
                int qi      = ws_q[b * K_N + k];
                float logit = verb_logits[((size_t)b * Q_N + qi) * V_N + v];
                float sig   = 1.f / (1.f + expf(-logit));
                float tv    = out_tv[b * K_N + k];
                out_fs[((size_t)b * K_N + k) * V_N + v] = sig * tv * acc.f[r];
            }
        }
    }
}

// =====================================================================
// Kernel 3: per-batch NMS (100 candidates) + apply keep to final scores.
// One workgroup (128 threads) per batch; everything in LDS (~6 KB).
// =====================================================================
__device__ __forceinline__ float iou_pp(const float* a, const float* c) {
    float areaA = (a[2] - a[0] + 1.f) * (a[3] - a[1] + 1.f);
    float areaB = (c[2] - c[0] + 1.f) * (c[3] - c[1] + 1.f);
    float xx1 = fmaxf(a[0], c[0]);
    float yy1 = fmaxf(a[1], c[1]);
    float xx2 = fminf(a[2], c[2]);
    float yy2 = fminf(a[3], c[3]);
    float w = fmaxf(0.f, xx2 - xx1 + 1.f);
    float h = fmaxf(0.f, yy2 - yy1 + 1.f);
    float inter = w * h;
    return inter / (areaA + areaB - inter);
}

__global__ __launch_bounds__(128)
void hoi_nms_kernel(float* __restrict__ out, const int* __restrict__ ws_lab)
{
    __shared__ float msc[K_N];
    __shared__ int   ord[K_N];
    __shared__ float bs[K_N][4];
    __shared__ float bo[K_N][4];
    __shared__ int   lab[K_N];
    __shared__ int   sup[K_N];
    __shared__ int   keep_s[K_N];
    __shared__ int   keep_o[K_N];

    const int b   = blockIdx.x;
    const int tid = threadIdx.x;
    float*       out_fs = out + OFF_FS + (size_t)b * K_N * V_N;
    const float* out_sb = out + OFF_SB + (size_t)b * K_N * 4;
    const float* out_ob = out + OFF_OB + (size_t)b * K_N * 4;
    float*       out_kp = out + OFF_KP + (size_t)b * K_N;

    for (int k = tid; k < K_N; k += 128) {
        float mx = -3.4e38f;
        for (int v = 0; v < V_N; ++v) mx = fmaxf(mx, out_fs[k * V_N + v]);
        msc[k] = mx;
        #pragma unroll
        for (int j = 0; j < 4; ++j) { bs[k][j] = out_sb[k * 4 + j]; bo[k][j] = out_ob[k * 4 + j]; }
        lab[k] = ws_lab[b * K_N + k];
        sup[k] = 0;
        keep_s[k] = 0;
    }
    __syncthreads();

    // Stable descending rank sort (matches argsort(-x) stability).
    for (int k = tid; k < K_N; k += 128) {
        int rnk = 0;
        float mk = msc[k];
        for (int j = 0; j < K_N; ++j) {
            float mj = msc[j];
            if (mj > mk || (mj == mk && j < k)) ++rnk;
        }
        ord[rnk] = k;
    }
    __syncthreads();

    // Greedy suppression in sorted order; dual-IoU computed on the fly.
    for (int i = 0; i < K_N; ++i) {
        int ki = !sup[i];                 // uniform across block
        if (tid == 0) keep_s[i] = ki;
        if (ki) {
            int oi = ord[i];
            for (int j = i + 1 + tid; j < K_N; j += 128) {
                int oj = ord[j];
                if (lab[oi] == lab[oj]) {
                    float ov = iou_pp(bs[oi], bs[oj]) * iou_pp(bo[oi], bo[oj]);
                    if (ov > NMS_TH) sup[j] = 1;
                }
            }
        }
        __syncthreads();
    }

    for (int i = tid; i < K_N; i += 128) keep_o[ord[i]] = keep_s[i];
    __syncthreads();

    for (int i = tid; i < K_N; i += 128) out_kp[i] = keep_o[i] ? 1.f : 0.f;
    for (int idx = tid; idx < K_N * V_N; idx += 128) {
        int k = idx / V_N;
        if (!keep_o[k]) out_fs[idx] = 0.f;
    }
}

// =====================================================================
// Launch
// =====================================================================
extern "C" void kernel_launch(void* const* d_in, const int* in_sizes, int n_in,
                              void* d_out, int out_size, void* d_ws, size_t ws_size,
                              hipStream_t stream)
{
    (void)in_sizes; (void)n_in; (void)out_size; (void)ws_size;

    const float* obj_logits  = (const float*)d_in[0];
    const float* verb_logits = (const float*)d_in[1];
    const float* sub_boxes   = (const float*)d_in[2];
    const float* obj_boxes   = (const float*)d_in[3];
    const float* correct_mat = (const float*)d_in[4];
    const int*   tsizes      = (const int*)d_in[5];
    float* out = (float*)d_out;

    int* ws_q   = (int*)d_ws;                      // B*K ints
    int* ws_lab = ws_q + (size_t)B_N * K_N;        // B*K ints (total ~200 KB)

    const size_t smem1 = (size_t)QC_N * sizeof(float)
                       + 1024 * sizeof(float) + 1024 * sizeof(int);

    hoi_softmax_topk_kernel<<<B_N, 1024, smem1, stream>>>(
        obj_logits, sub_boxes, obj_boxes, tsizes, out, ws_q, ws_lab);

    hoi_mask_score_kernel<<<B_N, 256, 0, stream>>>(
        verb_logits, correct_mat, ws_q, ws_lab, out);

    hoi_nms_kernel<<<B_N, 128, 0, stream>>>(out, ws_lab);
}
